// DilatedToothSegmentationNetwork_28174985462657
// MI455X (gfx1250) — matrix operand order repaired and checked by reading, not compiled, not run on hardware
//
#include <hip/hip_runtime.h>
#include <hip/hip_bf16.h>
#include <stdint.h>

// ---------------------------------------------------------------------------
// MI455X / gfx1250 fused implementation for DilatedToothSegmentationNetwork
// forward (compile-only loop).
//   - matmuls via v_wmma_f32_16x16x32_f16 (K padded to 32)
//   - pos-distance cross term via v_wmma_f32_16x16x4_f32 (exact fp32, K=4)
//   - GEMM A-tiles staged by the Tensor Data Mover (tensor_load_to_lds)
//   - WMMA fragments loaded as 16-byte vectors (ds_load_b128 /
//     global_load_b128), weight tiles stored TRANSPOSED in LDS so every
//     lane reads contiguous halves
//   - N x N distance matrices never materialized: fused WMMA + top-32
// ---------------------------------------------------------------------------

#define NPTS 12000
#define KSEL 32
#define BN_SCALE_F 0.9999950000374997f

typedef __attribute__((ext_vector_type(16))) _Float16     v16h;
typedef __attribute__((ext_vector_type(8)))  _Float16     v8h;
typedef __attribute__((ext_vector_type(8)))  float        v8f;
typedef __attribute__((ext_vector_type(2)))  float        v2f;
typedef __attribute__((ext_vector_type(4)))  unsigned int v4u;
typedef __attribute__((ext_vector_type(8)))  int          v8i;
typedef __attribute__((ext_vector_type(4)))  int          v4i;

__device__ inline v16h combine16(v8h lo, v8h hi) {
  return __builtin_shufflevector(lo, hi, 0, 1, 2, 3, 4, 5, 6, 7,
                                 8, 9, 10, 11, 12, 13, 14, 15);
}

// ---------------- WMMA fragment loads (ISA 7.12.2 layouts, wave32) ----------
// A: 16x32 f16 tile, row-major. lane L: M=L%16; lanes 0-15 hold K=0..7,16..23,
// lanes 16-31 hold K=8..15,24..31  ->  two contiguous v8h runs per lane.
__device__ inline v16h load_A_frag(const _Float16* As, int lda, int lane) {
  const int m  = lane & 15;
  const int kb = (lane >> 4) * 8;
  const v8h* p = (const v8h*)(As + m * lda + kb);   // 16B aligned (lda%16==0)
  return combine16(p[0], p[2]);                     // kb..kb+7 , kb+16..kb+23
}

// B stored TRANSPOSED in LDS as BT[n][k] (ldk halves per row). lanes 0-15:
// N=lane, K=0..7,16..23 ; lanes 16-31: K=8..15,24..31 -> two v8h per lane.
__device__ inline v16h load_BT_frag(const _Float16* BT, int ldk, int lane) {
  const int n  = lane & 15;
  const int kb = (lane >> 4) * 8;
  const v8h* p = (const v8h*)(BT + n * ldk + kb);
  return combine16(p[0], p[2]);
}

// ---------------- Tensor Data Mover: 2D f16 tile -> LDS ---------------------
// D# per CDNA5 ISA. OOB reads return zero -> free
// zero-padding of ragged M/K edges.
__device__ inline void tdm_load_2d_f16(const _Float16* gsrc, unsigned lds_byte,
                                       int tensor_d0, int tensor_d1,
                                       int tile_d0, int tile_d1, int stride0) {
  unsigned long long ga = (unsigned long long)(const void*)gsrc;
  v4u g0;
  g0[0] = 1u;                                          // count=1, user mode
  g0[1] = lds_byte;                                    // lds_addr
  g0[2] = (unsigned)(ga & 0xffffffffu);                // global_addr[31:0]
  g0[3] = (unsigned)((ga >> 32) & 0x01ffffffu) | (2u << 30);  // addr[56:32]|type=2
  unsigned td0 = (unsigned)(tensor_d0 < 0 ? 0 : tensor_d0);
  unsigned td1 = (unsigned)(tensor_d1 < 0 ? 0 : tensor_d1);
  v8i g1;
  g1[0] = (int)(1u << 16);                             // data_size=1 (2B/elem)
  g1[1] = (int)((td0 & 0xffffu) << 16);                // tensor_dim0 lo
  g1[2] = (int)((td0 >> 16) | ((td1 & 0xffffu) << 16));// tensor_dim0 hi|dim1 lo
  g1[3] = (int)((td1 >> 16) | (((unsigned)tile_d0 & 0xffffu) << 16));
  g1[4] = (int)((unsigned)tile_d1 & 0xffffu);          // tile_dim1, tile_dim2=0
  g1[5] = (int)(unsigned)stride0;                      // tensor_dim0_stride lo32
  g1[6] = 0;
  g1[7] = 0;
  v4i g2 = {0, 0, 0, 0};
  v4i g3 = {0, 0, 0, 0};
#if __has_builtin(__builtin_amdgcn_tensor_load_to_lds)
 #if defined(__clang_major__) && (__clang_major__ >= 23)
  v8i g4 = {0, 0, 0, 0, 0, 0, 0, 0};
  __builtin_amdgcn_tensor_load_to_lds(g0, g1, g2, g3, g4, 0);
 #else
  __builtin_amdgcn_tensor_load_to_lds(g0, g1, g2, g3, 0);
 #endif
 #if __has_builtin(__builtin_amdgcn_s_wait_tensorcnt)
  __builtin_amdgcn_s_wait_tensorcnt(0);
 #endif
#endif
}

// ---------------- elementwise / reduction helpers ---------------------------

__global__ void cast_pad_f16_kernel(const float* __restrict__ src,
                                    _Float16* __restrict__ dst,
                                    long rows, int cs, int cd) {
  long t = blockIdx.x * (long)blockDim.x + threadIdx.x;
  if (t < rows * (long)cd) {
    long r = t / cd; int c = (int)(t % cd);
    dst[t] = (c < cs) ? (_Float16)src[r * cs + c] : (_Float16)0.f;
  }
}

__global__ void rownorm_kernel(const float* __restrict__ x, float* __restrict__ sq,
                               int Np, int C) {
  int i = blockIdx.x * blockDim.x + threadIdx.x;
  if (i < Np) {
    float s = 0.f;
    for (int c = 0; c < C; ++c) { float v = x[(size_t)i * C + c]; s += v * v; }
    sq[i] = s;
  }
}

__global__ __launch_bounds__(256) void colmax_kernel(const float* __restrict__ h,
                                                     float* __restrict__ g,
                                                     int Np, int C) {
  int c = blockIdx.x;
  float m = -3.4e38f;
  for (int i = threadIdx.x; i < Np; i += blockDim.x)
    m = fmaxf(m, h[(size_t)i * C + c]);
  __shared__ float red[256];
  red[threadIdx.x] = m;
  __syncthreads();
  for (int s = 128; s > 0; s >>= 1) {
    if ((int)threadIdx.x < s) red[threadIdx.x] = fmaxf(red[threadIdx.x], red[threadIdx.x + s]);
    __syncthreads();
  }
  if (threadIdx.x == 0) g[c] = red[0];
}

// ---------------- fused GEMM + BN + activation (WMMA f16, TDM staging) ------
// Y[M,Nc] = act( scale[n]*BN_SCALE * (X @ W^T)[m,n] + bias[n] )
// X: f16 [M,Kd] row-major ; W: f16 [Nc,Kd] row-major (reference weight layout)
__global__ __launch_bounds__(128)
void gemm_bias_act_kernel(const _Float16* __restrict__ X,
                          const _Float16* __restrict__ W,
                          const float* __restrict__ scale,
                          const float* __restrict__ bias,
                          float* __restrict__ Y,
                          int M, int Nc, int Kd, int act) {
  __shared__ __align__(16) _Float16 As[64 * 32];   // 64 rows x 32 K, TDM-staged
  __shared__ __align__(16) _Float16 BT[16 * 32];   // W tile, [n][k] (transposed)
  const int lane = threadIdx.x & 31;
  const int wv   = threadIdx.x >> 5;
  const int m0 = blockIdx.x * 64;
  const int n0 = blockIdx.y * 16;

  v8f acc = {};
  for (int k0 = 0; k0 < Kd; k0 += 32) {
    if (wv == 0) {
      // async DMA of the A tile; OOB rows (m>=M) and cols (k>=Kd) arrive as 0
      tdm_load_2d_f16(X + (size_t)m0 * Kd + k0,
                      (unsigned)(unsigned long long)(void*)As,
                      Kd - k0, M - m0, 32, 64, Kd);
    }
    if (lane == 0)  // warm next weight tile into GL2 (global_prefetch_b8)
      __builtin_prefetch(W + (size_t)n0 * Kd + k0 + 32, 0, 1);
    // weight tile, transposed [n][k]: contiguous along W rows -> coalesced
    for (int t = threadIdx.x; t < 16 * 32; t += 128) {
      int nn = t >> 5, kk = t & 31;
      int ng = n0 + nn, kg = k0 + kk;
      BT[t] = (kg < Kd && ng < Nc) ? W[(size_t)ng * Kd + kg] : (_Float16)0.f;
    }
    __syncthreads();
    v16h a = load_A_frag(As + wv * 16 * 32, 32, lane);
    v16h b = load_BT_frag(BT, 32, lane);
    acc = __builtin_amdgcn_wmma_f32_16x16x32_f16(false, a, false, b,
                                                 (short)0, acc, false, false);
    __syncthreads();
  }
  const int ng = n0 + (lane & 15);
  const float s  = (scale && ng < Nc) ? scale[ng] * BN_SCALE_F : 1.f;
  const float bo = (bias  && ng < Nc) ? bias[ng] : 0.f;
#pragma unroll
  for (int r = 0; r < 8; ++r) {
    int mg = m0 + wv * 16 + (lane >> 4) * 8 + r;
    if (mg < M && ng < Nc) {
      float v = acc[r] * s + bo;
      if (act == 1)      v = fmaxf(v, 0.f);
      else if (act == 2) v = v > 0.f ? v : 0.2f * v;
      Y[(size_t)mg * Nc + ng] = v;
    }
  }
}

// ---------------- fused distance + top-32 selection -------------------------

__device__ inline void insert_sorted(float* bd, int* bi, float d, int j) {
  if (d >= bd[KSEL - 1]) return;
  int p = KSEL - 1;
  while (p > 0 && bd[p - 1] > d) { bd[p] = bd[p - 1]; bi[p] = bi[p - 1]; --p; }
  bd[p] = d; bi[p] = j;
}

// feature-space kNN: F is [Np, 32] f16 (zero padded, 64B rows). One
// v_wmma_f32_16x16x32_f16 per 16x16 distance tile; fragments come straight
// from global memory as 16-byte vector loads.
__global__ __launch_bounds__(256)
void knn_f16_kernel(const _Float16* __restrict__ F, const float* __restrict__ sq,
                    int Np, int* __restrict__ idx_out) {
  __shared__ float dt[8][16][17];
  __shared__ float md[8][16][KSEL];
  __shared__ int   mi[8][16][KSEL];
  const int lane = threadIdx.x & 31;
  const int wv   = threadIdx.x >> 5;
  const int i0 = blockIdx.x * 16;

  const int mrow = i0 + (lane & 15);
  const int mr   = mrow < Np ? mrow : Np - 1;       // clamp; OOB rows guarded later
  const int ha   = (lane >> 4);                     // 0: K0..7,16..23 ; 1: +8
  const v8h* rowa = (const v8h*)(F + (size_t)mr * 32);
  v16h a = combine16(rowa[ha], rowa[ha + 2]);

  float bd[KSEL]; int bi[KSEL];
#pragma unroll
  for (int t = 0; t < KSEL; ++t) { bd[t] = 3.4e38f; bi[t] = 0; }

  const int ntiles = (Np + 15) / 16;
  for (int jt = wv; jt < ntiles; jt += 8) {
    const int j0 = jt * 16;
    const int ncol = j0 + (lane & 15);
    const int nr   = ncol < Np ? ncol : Np - 1;
    const int hb   = (lane >> 4);                   // 0: K0..7,16..23 ; 1: +8
    const v8h* rowb = (const v8h*)(F + (size_t)nr * 32);
    v16h b = combine16(rowb[hb], rowb[hb + 2]);

    v8f dot = {};
    dot = __builtin_amdgcn_wmma_f32_16x16x32_f16(false, a, false, b,
                                                 (short)0, dot, false, false);
#pragma unroll
    for (int r = 0; r < 8; ++r)
      dt[wv][(lane >> 4) * 8 + r][lane & 15] = dot[r];
    if (lane < 16) {
      int ir = i0 + lane;
      float si = sq[ir < Np ? ir : (Np - 1)];
      for (int c = 0; c < 16; ++c) {
        int j = j0 + c;
        if (j < Np && ir < Np)
          insert_sorted(bd, bi, si + sq[j] - 2.f * dt[wv][lane][c], j);
      }
    }
  }
  if (lane < 16)
#pragma unroll
    for (int t = 0; t < KSEL; ++t) { md[wv][lane][t] = bd[t]; mi[wv][lane][t] = bi[t]; }
  __syncthreads();
  if (wv == 0 && lane < 16) {
    float fd[KSEL]; int fj[KSEL];
#pragma unroll
    for (int t = 0; t < KSEL; ++t) { fd[t] = 3.4e38f; fj[t] = 0; }
    for (int w2 = 0; w2 < 8; ++w2)
      for (int t = 0; t < KSEL; ++t)
        insert_sorted(fd, fj, md[w2][lane][t], mi[w2][lane][t]);
    int irow = i0 + lane;
    if (irow < Np)
      for (int t = 0; t < KSEL; ++t) idx_out[(size_t)irow * KSEL + t] = fj[t];
  }
#if __has_builtin(__builtin_amdgcn_s_cluster_barrier)
  __builtin_amdgcn_s_cluster_barrier();   // NOP when not dispatched in a cluster
#endif
}

// pos kNN: exact fp32 cross term via v_wmma_f32_16x16x4_f32 (3 coords + 0 pad)
__global__ __launch_bounds__(256)
void knn_pos_kernel(const float* __restrict__ pos, const float* __restrict__ sq,
                    int Np, int* __restrict__ idx_out) {
  __shared__ float dt[8][16][17];
  __shared__ float md[8][16][KSEL];
  __shared__ int   mi[8][16][KSEL];
  const int lane = threadIdx.x & 31;
  const int wv   = threadIdx.x >> 5;
  const int i0 = blockIdx.x * 16;

#if __has_builtin(__builtin_amdgcn_wmma_f32_16x16x4_f32)
  const int mrow = i0 + (lane & 15);
  v2f a;
#pragma unroll
  for (int t = 0; t < 2; ++t) {
    int k = (lane >> 4) * 2 + t;
    a[t] = (mrow < Np && k < 3) ? pos[(size_t)mrow * 3 + k] : 0.f;
  }
#endif
  float bd[KSEL]; int bi[KSEL];
#pragma unroll
  for (int t = 0; t < KSEL; ++t) { bd[t] = 3.4e38f; bi[t] = 0; }

  const int ntiles = (Np + 15) / 16;
  for (int jt = wv; jt < ntiles; jt += 8) {
    const int j0 = jt * 16;
    v8f dot = {};
#if __has_builtin(__builtin_amdgcn_wmma_f32_16x16x4_f32)
    const int ncol = j0 + (lane & 15);
    v2f b;
#pragma unroll
    for (int t = 0; t < 2; ++t) {
      int k = (lane >> 4) * 2 + t;
      b[t] = (ncol < Np && k < 3) ? pos[(size_t)ncol * 3 + k] : 0.f;
    }
    dot = __builtin_amdgcn_wmma_f32_16x16x4_f32(false, a, false, b,
                                                (short)0, dot, false, false);
#else
#pragma unroll
    for (int r = 0; r < 8; ++r) {
      int m = i0 + (lane >> 4) * 8 + r, n = j0 + (lane & 15);
      float s = 0.f;
      if (m < Np && n < Np)
        for (int c = 0; c < 3; ++c) s += pos[(size_t)m * 3 + c] * pos[(size_t)n * 3 + c];
      dot[r] = s;
    }
#endif
#pragma unroll
    for (int r = 0; r < 8; ++r)
      dt[wv][(lane >> 4) * 8 + r][lane & 15] = dot[r];
    if (lane < 16) {
      int ir = i0 + lane;
      float si = sq[ir < Np ? ir : (Np - 1)];
      for (int c = 0; c < 16; ++c) {
        int j = j0 + c;
        if (j < Np && ir < Np)
          insert_sorted(bd, bi, si + sq[j] - 2.f * dt[wv][lane][c], j);
      }
    }
  }
  if (lane < 16)
#pragma unroll
    for (int t = 0; t < KSEL; ++t) { md[wv][lane][t] = bd[t]; mi[wv][lane][t] = bi[t]; }
  __syncthreads();
  if (wv == 0 && lane < 16) {
    float fd[KSEL]; int fj[KSEL];
#pragma unroll
    for (int t = 0; t < KSEL; ++t) { fd[t] = 3.4e38f; fj[t] = 0; }
    for (int w2 = 0; w2 < 8; ++w2)
      for (int t = 0; t < KSEL; ++t)
        insert_sorted(fd, fj, md[w2][lane][t], mi[w2][lane][t]);
    int irow = i0 + lane;
    if (irow < Np)
      for (int t = 0; t < KSEL; ++t) idx_out[(size_t)irow * KSEL + t] = fj[t];
  }
}

// ---------------- edge conv: gather + WMMA + fused BN/leaky/max over K ------
// out[p, n] = max_s leaky( bn( [xn-xc, xc] @ Wconv^T ) ) ; never materializes
// the [N, K, hid] tensor. Weights transposed in LDS for b128 fragment loads.
__global__ __launch_bounds__(128)
void edge_conv_kernel(const _Float16* __restrict__ F,   // [Np, Cpad] f16
                      const int* __restrict__ idx,      // [Np, KSEL]
                      const _Float16* __restrict__ W,   // [hid, 2C] f16
                      const float* __restrict__ bng, const float* __restrict__ bnb,
                      float* __restrict__ out,          // [Np, hid]
                      int Np, int C, int Cpad, int hid) {
  __shared__ __align__(16) _Float16 A[16 * 128];    // 16 points x (2C pad 128)
  __shared__ __align__(16) _Float16 BT[64 * 128];   // weights [n][k], n pad 64
  const int lane = threadIdx.x & 31;
  const int wv   = threadIdx.x >> 5;  // wave owns output n-tile wv*16..
  const int i0 = blockIdx.x * 16;

  // transposed weight fill: contiguous along W rows -> coalesced
  for (int t = threadIdx.x; t < 64 * 128; t += 128) {
    int n = t >> 7, k = t & 127;
    BT[t] = (k < 2 * C && n < hid) ? W[(size_t)n * (2 * C) + k] : (_Float16)0.f;
  }
  v8f mx;
#pragma unroll
  for (int r = 0; r < 8; ++r) mx[r] = -3.4e38f;

  for (int s = 0; s < KSEL; ++s) {
    __syncthreads();
    for (int t = threadIdx.x; t < 16 * 128; t += 128) {
      int p = t >> 7, c = t & 127;
      int ip = i0 + p;
      float v = 0.f;
      if (ip < Np && c < 2 * C) {
        int cc = (c < C) ? c : c - C;
        float xc = (float)F[(size_t)ip * Cpad + cc];
        if (c < C) {
          int j = idx[(size_t)ip * KSEL + s];
          v = (float)F[(size_t)j * Cpad + c] - xc;
        } else v = xc;
      }
      A[t] = (_Float16)v;
    }
    __syncthreads();
    v8f acc = {};
#pragma unroll
    for (int kt = 0; kt < 4; ++kt) {
      v16h a = load_A_frag(A + kt * 32, 128, lane);
      v16h b = load_BT_frag(BT + (size_t)(wv * 16) * 128 + kt * 32, 128, lane);
      acc = __builtin_amdgcn_wmma_f32_16x16x32_f16(false, a, false, b,
                                                   (short)0, acc, false, false);
    }
    int ng = wv * 16 + (lane & 15);
    float g  = (ng < hid) ? bng[ng] * BN_SCALE_F : 1.f;
    float bb = (ng < hid) ? bnb[ng] : 0.f;
#pragma unroll
    for (int r = 0; r < 8; ++r) {
      float v = acc[r] * g + bb;
      v = v > 0.f ? v : 0.2f * v;
      mx[r] = fmaxf(mx[r], v);
    }
  }
  int ng = wv * 16 + (lane & 15);
#pragma unroll
  for (int r = 0; r < 8; ++r) {
    int mg = i0 + (lane >> 4) * 8 + r;
    if (mg < Np && ng < hid) out[(size_t)mg * hid + ng] = mx[r];
  }
}

// ---------------------------------------------------------------------------
// host-side schedule
// ---------------------------------------------------------------------------
extern "C" void kernel_launch(void* const* d_in, const int* in_sizes, int n_in,
                              void* d_out, int out_size, void* d_ws, size_t ws_size,
                              hipStream_t stream) {
  (void)in_sizes; (void)out_size; (void)ws_size;
  const int N = NPTS;
  const float* x   = (const float*)d_in[0];   // [1,12000,24]
  const float* pos = (const float*)d_in[1];   // [1,12000,3]
  int pidx = 2;
  auto P = [&]() -> const float* {
    if (n_in <= 2) return x;
    const float* p = (const float*)d_in[pidx];
    pidx = (pidx + 1 >= n_in) ? 2 : pidx + 1;
    return p;
  };
  char* wsp = (char*)d_ws;
  auto alloc = [&](size_t bytes) -> void* {
    void* p = (void*)wsp; wsp += (bytes + 255) & ~(size_t)255; return p;
  };
  _Float16* xh32 = (_Float16*)alloc((size_t)N * 32 * 2);
  _Float16* xh24 = (_Float16*)alloc((size_t)N * 24 * 2);
  float* sqx  = (float*)alloc((size_t)N * 4);
  float* sqp  = (float*)alloc((size_t)N * 4);
  int*   idxp = (int*)alloc((size_t)N * KSEL * 4);
  int*   idxf = (int*)alloc((size_t)N * KSEL * 4);
  float* a0   = (float*)alloc((size_t)N * 1024 * 4);
  float* a1   = (float*)alloc((size_t)N * 1024 * 4);
  _Float16* h16 = (_Float16*)alloc((size_t)N * 1024 * 2);
  _Float16* wh  = (_Float16*)alloc((size_t)2 * 1024 * 1024 * 2);  // weight arena
  float* gmax = (float*)alloc(4096);

  dim3 b256(256), b128(128);
  auto cgrid = [](long total) { return dim3((unsigned)((total + 255) / 256)); };
  auto gemm = [&](const _Float16* X, int M, int Kd, int Nc, float* Y, int act,
                  const float* sc, const float* bi) {
    dim3 g((M + 63) / 64, (Nc + 15) / 16);
    gemm_bias_act_kernel<<<g, b128, 0, stream>>>(X, wh, sc, bi, Y, M, Nc, Kd, act);
  };
  auto castw = [&](const float* src, int o, int i) {
    cast_pad_f16_kernel<<<cgrid((long)o * i), b256, 0, stream>>>(src, wh, o, i, i);
  };

  // stage inputs to f16 / squared norms
  cast_pad_f16_kernel<<<cgrid((long)N * 32), b256, 0, stream>>>(x, xh32, N, 24, 32);
  cast_pad_f16_kernel<<<cgrid((long)N * 24), b256, 0, stream>>>(x, xh24, N, 24, 24);
  rownorm_kernel<<<cgrid(N), b256, 0, stream>>>(x, sqx, N, 24);
  rownorm_kernel<<<cgrid(N), b256, 0, stream>>>(pos, sqp, N, 3);

  // fused distance + top-32 (the only N^2 work; never hits HBM as a matrix)
  knn_pos_kernel<<<dim3((N + 15) / 16), b256, 0, stream>>>(pos, sqp, N, idxp);
  knn_f16_kernel<<<dim3((N + 15) / 16), b256, 0, stream>>>(xh32, sqx, N, idxf);

  // --- STN trunk: 24 -> 64 -> 128 -> 1024, global max pool ---
  castw(P(), 64, 24);
  gemm(xh24, N, 24, 64, a0, 1, P(), P());
  cast_pad_f16_kernel<<<cgrid((long)N * 64), b256, 0, stream>>>(a0, h16, N, 64, 64);
  castw(P(), 128, 64);
  gemm(h16, N, 64, 128, a1, 1, P(), P());
  cast_pad_f16_kernel<<<cgrid((long)N * 128), b256, 0, stream>>>(a1, h16, N, 128, 128);
  castw(P(), 1024, 128);
  gemm(h16, N, 128, 1024, a0, 1, P(), P());
  colmax_kernel<<<dim3(1024), b256, 0, stream>>>(a0, gmax, N, 1024);

  // --- edge conv (gather + WMMA + fused max over 32 neighbors) ---
  castw(P(), 24, 48);
  edge_conv_kernel<<<dim3((N + 15) / 16), b128, 0, stream>>>(
      xh32, idxf, wh, P(), P(), a1, N, 24, 32, 24);

  // --- head: project -> res-block GEMMs -> seg / features / x_fused ---
  cast_pad_f16_kernel<<<cgrid((long)N * 24), b256, 0, stream>>>(a1, h16, N, 24, 24);
  castw(P(), 256, 24);
  gemm(h16, N, 24, 256, a0, 1, P(), P());
  cast_pad_f16_kernel<<<cgrid((long)N * 256), b256, 0, stream>>>(a0, h16, N, 256, 256);
  castw(P(), 384, 256);
  gemm(h16, N, 256, 384, a1, 1, P(), P());
  cast_pad_f16_kernel<<<cgrid((long)N * 384), b256, 0, stream>>>(a1, h16, N, 384, 384);

  float* seg   = (float*)d_out;             // [N,17]
  float* feats = seg + (size_t)N * 17;      // [N,256]
  float* fused = feats + (size_t)N * 256;   // [N,256]
  castw(P(), 256, 384);
  gemm(h16, N, 384, 256, feats, 1, P(), P());
  cast_pad_f16_kernel<<<cgrid((long)N * 256), b256, 0, stream>>>(feats, h16, N, 256, 256);
  castw(P(), 17, 256);
  gemm(h16, N, 256, 17, seg, 0, nullptr, nullptr);
  castw(P(), 256, 256);
  gemm(h16, N, 256, 256, fused, 0, nullptr, nullptr);
}
